// WindowAttention_22101901705461
// MI455X (gfx1250) — matrix operand order repaired and checked by reading, not compiled, not run on hardware
//
#include <hip/hip_runtime.h>

// ---------------------------------------------------------------------------
// Swin-v2 window attention for MI455X (gfx1250), bf16 WMMA w/ f32 accumulate.
// DIM=256, WIN=64, HEADS=8, head_dim=32, 2048 windows, 256 masks.
// ---------------------------------------------------------------------------

#define DIMC   256
#define WINN   64
#define NHEAD  8
#define HD     32
#define NWIN   256
#define NBATCH 2048

// LDS row strides (padded by 8 ushorts = 16B to avoid 64-bank conflicts on
// stride-row fragment loads: 264 ushorts = 132 dwords, 132 % 64 = 4).
#define XS 264   // rows of X / Q / K (256 cols + pad)
#define VS 72    // rows of V^T and P strips (64 cols + pad)

typedef __attribute__((ext_vector_type(16))) __bf16 v16bf;
typedef __attribute__((ext_vector_type(8)))  float  v8f;

union Frag { uint4 q[2]; v16bf v; };

__device__ __forceinline__ unsigned short f2bf(float f) {
    unsigned u = __float_as_uint(f);
    return (unsigned short)((u + 0x7FFFu + ((u >> 16) & 1u)) >> 16);  // RNE
}
__device__ __forceinline__ float bf2f(unsigned short h) {
    return __uint_as_float(((unsigned)h) << 16);
}

__device__ __forceinline__ v16bf ld_frag(const unsigned short* a,
                                         const unsigned short* b) {
    Frag f;
    f.q[0] = *(const uint4*)a;
    f.q[1] = *(const uint4*)b;
    return f.v;
}

__device__ __forceinline__ v8f wmma_bf16(v16bf a, v16bf b, v8f c) {
    return __builtin_amdgcn_wmma_f32_16x16x32_bf16(
        false, a, false, b, (short)0, c, false, false);
}

// ---------------------------------------------------------------------------
// Prep: bf16 weight copies, dense relative-position bias, clamped logit scale
// ---------------------------------------------------------------------------
__global__ void winattn_prep(const float* __restrict__ qkv_w,
                             const float* __restrict__ qkv_b,
                             const float* __restrict__ proj_w,
                             const float* __restrict__ proj_b,
                             const float* __restrict__ logit_scale,
                             const float* __restrict__ rel_tab,
                             const int*   __restrict__ rpi,
                             unsigned short* __restrict__ qkvw_bf,
                             unsigned short* __restrict__ projw_bf,
                             float* __restrict__ qkvb_f,
                             float* __restrict__ projb_f,
                             float* __restrict__ bias_f,
                             float* __restrict__ ls_f) {
    const int NQW = 3 * DIMC * DIMC;     // 196608
    const int NPW = DIMC * DIMC;         // 65536
    const int NQB = 3 * DIMC;            // 768
    const int NPB = DIMC;                // 256
    const int NBI = NHEAD * WINN * WINN; // 32768
    const int NT  = NQW + NPW + NQB + NPB + NBI + NHEAD;
    for (int i = blockIdx.x * blockDim.x + threadIdx.x; i < NT;
         i += gridDim.x * blockDim.x) {
        int t = i;
        if (t < NQW) { qkvw_bf[t] = f2bf(qkv_w[t]); continue; }
        t -= NQW;
        if (t < NPW) { projw_bf[t] = f2bf(proj_w[t]); continue; }
        t -= NPW;
        if (t < NQB) { qkvb_f[t] = qkv_b[t]; continue; }
        t -= NQB;
        if (t < NPB) { projb_f[t] = proj_b[t]; continue; }
        t -= NPB;
        if (t < NBI) {
            int h  = t >> 12;        // / 4096
            int rc = t & 4095;       // n*64 + m
            bias_f[t] = rel_tab[rpi[rc] * NHEAD + h];
            continue;
        }
        t -= NBI;
        ls_f[t] = __expf(fminf(logit_scale[t], 4.605170185988092f)); // log(100)
    }
}

// ---------------------------------------------------------------------------
// Main fused kernel: 1 block = 1 window, 256 threads = 8 waves
// ---------------------------------------------------------------------------
__global__ __launch_bounds__(256) void winattn_main(
    const float* __restrict__ x, const float* __restrict__ mask,
    const unsigned short* __restrict__ qkvw_bf,
    const unsigned short* __restrict__ projw_bf,
    const float* __restrict__ qkvb_f, const float* __restrict__ projb_f,
    const float* __restrict__ bias_f, const float* __restrict__ ls_f,
    float* __restrict__ out) {
    const int b    = blockIdx.x;
    const int tid  = threadIdx.x;
    const int lane = tid & 31;
    const int wave = tid >> 5;
    const int hl   = (lane >> 4) & 1;  // lane half (K-group select)
    const int ln   = lane & 15;

    extern __shared__ unsigned short smem[];
    unsigned short* Xs = smem;                 // [64][XS]  x, later O (bf16)
    unsigned short* Qs = Xs + WINN * XS;       // [64][XS]  q (normalized*ls)
    unsigned short* Ks = Qs + WINN * XS;       // [64][XS]  k (normalized)
    unsigned short* Vt = Ks + WINN * XS;       // [256][VS] v transposed
    unsigned short* Pb = Vt + DIMC * VS;       // [8][16][VS] per-wave P strip

    // ---- Phase A: load x tile, convert f32 -> bf16 into padded LDS --------
    const float* xb = x + (size_t)b * (WINN * DIMC);
    for (int i = tid; i < WINN * DIMC / 4; i += 256) {
        float4 f = ((const float4*)xb)[i];
        int row = i >> 6, col = (i & 63) * 4;
        unsigned short* d = Xs + row * XS + col;
        d[0] = f2bf(f.x); d[1] = f2bf(f.y); d[2] = f2bf(f.z); d[3] = f2bf(f.w);
    }
    __syncthreads();

    // ---- Phase B: QKV = X(64x256) @ W^T (+bias); 192 tiles over 8 waves ---
    for (int t = wave; t < 192; t += 8) {
        int mt = t & 3;           // row-tile 0..3
        int oc = t >> 2;          // col-tile 0..47
        int o  = oc * 16 + ln;    // this lane's output column (0..767)
        v8f acc = {};
        const unsigned short* arow = Xs + (mt * 16 + ln) * XS;
        const unsigned short* brow = qkvw_bf + o * DIMC;
#pragma unroll
        for (int kt = 0; kt < 8; ++kt) {
            int kb = kt * 32;
            v16bf A = ld_frag(arow + kb + 8 * hl, arow + kb + 16 + 8 * hl);
            v16bf B = ld_frag(brow + kb + 8 * hl, brow + kb + 16 + 8 * hl);
            acc = wmma_bf16(A, B, acc);
        }
        float bias = qkvb_f[o];
        int r0 = mt * 16 + 8 * hl;
        if (o < 512) {            // Q or K, row-major [n][c]
            unsigned short* dst = (o < 256) ? (Qs + o) : (Ks + (o - 256));
#pragma unroll
            for (int v = 0; v < 8; ++v)
                dst[(r0 + v) * XS] = f2bf(acc[v] + bias);
        } else {                  // V stored transposed: Vt[d][m]
            alignas(16) unsigned short tmp[8];
#pragma unroll
            for (int v = 0; v < 8; ++v) tmp[v] = f2bf(acc[v] + bias);
            *(uint4*)(Vt + (o - 512) * VS + r0) = *(const uint4*)tmp;
        }
    }
    __syncthreads();

    // ---- Phase C: L2-normalize q (fold logit scale) and k, per head-row ---
    for (int t = tid; t < 2 * NHEAD * WINN; t += 256) {
        int which = t >> 9;            // 0 = q, 1 = k
        int h = (t >> 6) & 7;
        int n = t & 63;
        unsigned short* p = (which ? Ks : Qs) + n * XS + h * HD;
        float vals[HD];
        float ss = 0.f;
#pragma unroll
        for (int d = 0; d < HD; ++d) { float f = bf2f(p[d]); vals[d] = f; ss += f * f; }
        float scale = 1.0f / fmaxf(sqrtf(ss), 1e-12f);
        if (!which) scale *= ls_f[h];
#pragma unroll
        for (int d = 0; d < HD; ++d) p[d] = f2bf(vals[d] * scale);
    }
    __syncthreads();

    // ---- Phase D: attention, one wave per head ----------------------------
    {
        const int h = wave;
        const float* biasH = bias_f + h * (WINN * WINN);
        const float* maskW = mask + (size_t)(b & (NWIN - 1)) * (WINN * WINN);
        unsigned short* P = Pb + wave * (16 * VS);
        for (int mt = 0; mt < 4; ++mt) {
            // S strip: 4 tiles of 16x16, each a single K=32 WMMA
            v8f s[4];
            const unsigned short* qrow = Qs + (mt * 16 + ln) * XS + h * HD;
            v16bf A = ld_frag(qrow + 8 * hl, qrow + 16 + 8 * hl);
#pragma unroll
            for (int jt = 0; jt < 4; ++jt) {
                const unsigned short* krow = Ks + (jt * 16 + ln) * XS + h * HD;
                v16bf B = ld_frag(krow + 8 * hl, krow + 16 + 8 * hl);
                v8f c = {};
                s[jt] = wmma_bf16(A, B, c);
            }
            // + relative-position bias + shifted-window mask
#pragma unroll
            for (int jt = 0; jt < 4; ++jt)
#pragma unroll
                for (int v = 0; v < 8; ++v) {
                    int n = mt * 16 + v + 8 * hl;
                    int m = jt * 16 + ln;
                    s[jt][v] += biasH[n * 64 + m] + maskW[n * 64 + m];
                }
            // Row softmax: rows live across lanes 0..15 / 16..31 per VGPR
#pragma unroll
            for (int v = 0; v < 8; ++v) {
                float mx = fmaxf(fmaxf(s[0][v], s[1][v]), fmaxf(s[2][v], s[3][v]));
#pragma unroll
                for (int d = 8; d >= 1; d >>= 1) mx = fmaxf(mx, __shfl_xor(mx, d, 32));
                float e0 = __expf(s[0][v] - mx), e1 = __expf(s[1][v] - mx);
                float e2 = __expf(s[2][v] - mx), e3 = __expf(s[3][v] - mx);
                float sm = e0 + e1 + e2 + e3;
#pragma unroll
                for (int d = 8; d >= 1; d >>= 1) sm += __shfl_xor(sm, d, 32);
                float inv = 1.0f / sm;
                s[0][v] = e0 * inv; s[1][v] = e1 * inv;
                s[2][v] = e2 * inv; s[3][v] = e3 * inv;
            }
            // Stage P strip (16x64 bf16) for A-operand layout reload
#pragma unroll
            for (int jt = 0; jt < 4; ++jt)
#pragma unroll
                for (int v = 0; v < 8; ++v)
                    P[(v + 8 * hl) * VS + jt * 16 + ln] = f2bf(s[jt][v]);
            // O strip = P(16x64) @ V(64x32): 2 col-tiles x 2 K-chunks
#pragma unroll
            for (int ot = 0; ot < 2; ++ot) {
                v8f acc = {};
#pragma unroll
                for (int kt = 0; kt < 2; ++kt) {
                    const unsigned short* prow = P + ln * VS + kt * 32;
                    v16bf Ap = ld_frag(prow + 8 * hl, prow + 16 + 8 * hl);
                    const unsigned short* vrow =
                        Vt + (h * HD + ot * 16 + ln) * VS + kt * 32;
                    v16bf Bv = ld_frag(vrow + 8 * hl, vrow + 16 + 8 * hl);
                    acc = wmma_bf16(Ap, Bv, acc);
                }
#pragma unroll
                for (int v = 0; v < 8; ++v) {
                    int row = mt * 16 + v + 8 * hl;
                    int col = h * HD + ot * 16 + ln;
                    Xs[row * XS + col] = f2bf(acc[v]);   // O overwrites x
                }
            }
        }
    }
    __syncthreads();

    // ---- Phase E: out = O(64x256) @ projW^T + proj_b ----------------------
    float* og = out + (size_t)b * (WINN * DIMC);
    for (int t = wave; t < 64; t += 8) {
        int mt = t & 3;
        int oc = t >> 2;          // 0..15
        int o  = oc * 16 + ln;
        v8f acc = {};
        const unsigned short* arow = Xs + (mt * 16 + ln) * XS;
        const unsigned short* brow = projw_bf + o * DIMC;
#pragma unroll
        for (int kt = 0; kt < 8; ++kt) {
            int kb = kt * 32;
            v16bf A = ld_frag(arow + kb + 8 * hl, arow + kb + 16 + 8 * hl);
            v16bf B = ld_frag(brow + kb + 8 * hl, brow + kb + 16 + 8 * hl);
            acc = wmma_bf16(A, B, acc);
        }
        float bias = projb_f[o];
#pragma unroll
        for (int v = 0; v < 8; ++v) {
            int row = mt * 16 + v + 8 * hl;
            og[row * DIMC + o] = acc[v] + bias;
        }
    }
}

// ---------------------------------------------------------------------------
extern "C" void kernel_launch(void* const* d_in, const int* in_sizes, int n_in,
                              void* d_out, int out_size, void* d_ws,
                              size_t ws_size, hipStream_t stream) {
    const float* x           = (const float*)d_in[0];
    const float* mask        = (const float*)d_in[1];
    const float* qkv_w       = (const float*)d_in[2];
    const float* qkv_b       = (const float*)d_in[3];
    const float* proj_w      = (const float*)d_in[4];
    const float* proj_b      = (const float*)d_in[5];
    const float* logit_scale = (const float*)d_in[6];
    const float* rel_tab     = (const float*)d_in[7];
    const int*   rpi         = (const int*)d_in[8];

    // Workspace layout
    char* ws = (char*)d_ws;
    unsigned short* qkvw_bf = (unsigned short*)ws;          // 196608 ushort
    unsigned short* projw_bf = qkvw_bf + 3 * DIMC * DIMC;   // 65536 ushort
    float* fbase  = (float*)(ws + (3 * DIMC * DIMC + DIMC * DIMC) * 2);
    float* qkvb_f = fbase;                 // 768
    float* projb_f = qkvb_f + 3 * DIMC;    // 256
    float* bias_f = projb_f + DIMC;        // 32768
    float* ls_f   = bias_f + NHEAD * WINN * WINN;  // 8

    winattn_prep<<<580, 256, 0, stream>>>(qkv_w, qkv_b, proj_w, proj_b,
                                          logit_scale, rel_tab, rpi, qkvw_bf,
                                          projw_bf, qkvb_f, projb_f, bias_f,
                                          ls_f);

    size_t shmem = (size_t)(3 * WINN * XS + DIMC * VS + NHEAD * 16 * VS) *
                   sizeof(unsigned short);  // 156672 bytes
    hipFuncSetAttribute((const void*)winattn_main,
                        hipFuncAttributeMaxDynamicSharedMemorySize, (int)shmem);
    winattn_main<<<NBATCH, 256, shmem, stream>>>(
        x, mask, qkvw_bf, projw_bf, qkvb_f, projb_f, bias_f, ls_f,
        (float*)d_out);
}
